// SharedWeightBiLSTM_18631568130376
// MI455X (gfx1250) — matrix operand order repaired and checked by reading, not compiled, not run on hardware
//
#include <hip/hip_runtime.h>
#include <hip/hip_bf16.h>
#include <stdint.h>

// ---------------------------------------------------------------------------
// Shared-weight bidirectional LSTM for MI455X (gfx1250, wave32, WMMA).
//
// Strategy:
//  * fp32 GEMMs executed on v_wmma_f32_16x16x32_bf16 via hi/lo bf16 split
//    (3 WMMA terms: Ahi*Bhi + Ahi*Blo + Alo*Bhi) with f32 accumulation.
//  * Per time step, per direction: z = [x_s | h] @ [Wi;Wh]^T + b  (K=1024).
//  * One persistent kernel runs all 4 layers x 256 steps for both directions,
//    synchronizing with a software grid barrier (32 co-resident workgroups).
//  * Weights are pre-swizzled into the exact wave32 WMMA B-operand layout so
//    each lane issues two contiguous global_load_b128 per fragment; A tiles
//    are staged in LDS and read with the ISA A-fragment swizzle (ds_load_b128).
//  * Cell state c for each 16x16 tile lives in 8 VGPRs for the whole scan.
// ---------------------------------------------------------------------------

#define S_LEN   256
#define BATCH   64
#define EMB     512
#define HID     512
#define NLAYER  4
#define GATES   2048            // 4*HID, gate order (i,f,o,g)
#define KDIM    1024            // EMB + HID
#define KB_CNT  32              // KDIM / 32
#define NT_CNT  128             // GATES / 16
#define NWG     32              // persistent workgroups (co-resident)
#define TPB     256             // 8 waves of 32

typedef __attribute__((ext_vector_type(16))) __bf16 v16bf;
typedef __attribute__((ext_vector_type(8)))  float  v8f;

union Frag16 { uint4 q[2]; v16bf v; };

// ---- scratch layout (bytes) ------------------------------------------------
static constexpr size_t WP_ELEMS  = (size_t)NLAYER * NT_CNT * KB_CNT * 512; // 8,388,608 bf16
static constexpr size_t XIN_ELEMS = (size_t)S_LEN * BATCH * EMB;            // 8,388,608
static constexpr size_t HS_ELEMS  = (size_t)S_LEN * BATCH * HID;            // 8,388,608

static constexpr size_t OFF_WP_HI  = 0;
static constexpr size_t OFF_WP_LO  = OFF_WP_HI  + WP_ELEMS  * 2;
static constexpr size_t OFF_XIN_HI = OFF_WP_LO  + WP_ELEMS  * 2;
static constexpr size_t OFF_XIN_LO = OFF_XIN_HI + XIN_ELEMS * 2;
static constexpr size_t OFF_HS_F   = OFF_XIN_LO + XIN_ELEMS * 2;
static constexpr size_t OFF_HS_B   = OFF_HS_F   + HS_ELEMS  * 4;
static constexpr size_t OFF_HSP    = OFF_HS_B   + HS_ELEMS  * 4;            // 4 * 64*512 bf16
static constexpr size_t OFF_BAR    = OFF_HSP    + (size_t)4 * BATCH * HID * 2;
// total ~134.5 MB of d_ws

// ---- grid barrier (sense/generation, device scope) -------------------------
__device__ __forceinline__ void gbar(unsigned* bar) {
  __syncthreads();
  if (threadIdx.x == 0) {
    unsigned gen = __hip_atomic_load(bar + 1, __ATOMIC_RELAXED, __HIP_MEMORY_SCOPE_AGENT);
    unsigned arr = __hip_atomic_fetch_add(bar, 1u, __ATOMIC_ACQ_REL, __HIP_MEMORY_SCOPE_AGENT);
    if (arr == NWG - 1) {
      __hip_atomic_store(bar, 0u, __ATOMIC_RELAXED, __HIP_MEMORY_SCOPE_AGENT);
      __hip_atomic_fetch_add(bar + 1, 1u, __ATOMIC_ACQ_REL, __HIP_MEMORY_SCOPE_AGENT);
    } else {
      while (__hip_atomic_load(bar + 1, __ATOMIC_ACQUIRE, __HIP_MEMORY_SCOPE_AGENT) == gen)
        __builtin_amdgcn_s_sleep(1);
    }
  }
  __syncthreads();
}

__device__ __forceinline__ float sigf(float x)  { return 1.0f / (1.0f + __expf(-x)); }
__device__ __forceinline__ float tanhfast(float x) {
  float e = __expf(2.0f * x);                 // inf-safe: x>>0 -> 1, x<<0 -> -1
  return 1.0f - 2.0f / (e + 1.0f);
}

// ---------------------------------------------------------------------------
// Barrier init (reset every call -> deterministic across graph replays)
// ---------------------------------------------------------------------------
__global__ void init_bar_kernel(unsigned* bar) { bar[0] = 0u; bar[1] = 0u; }

// ---------------------------------------------------------------------------
// Weight prep: split W_cat = [Wi | Wh] per output column into bf16 hi/lo and
// swizzle into the wave32 WMMA B-operand layout:
//   block (l, nt, kb) = 512 bf16 = 32 lanes x 16 values; lane<16 -> local K
//   {0..7,16..23}, lane>=16 -> local K {8..15,24..31}; column n = nt*16 + (lane&15).
// ---------------------------------------------------------------------------
__global__ void prep_weights_kernel(const float* __restrict__ Wi,
                                    const float* __restrict__ Wh,
                                    __bf16* __restrict__ wpH,
                                    __bf16* __restrict__ wpL) {
  const size_t total  = WP_ELEMS;
  const size_t stride = (size_t)gridDim.x * blockDim.x;
  for (size_t idx = (size_t)blockIdx.x * blockDim.x + threadIdx.x; idx < total; idx += stride) {
    unsigned q    = (unsigned)(idx        & 15u);
    unsigned lane = (unsigned)((idx >> 4) & 31u);
    unsigned kb   = (unsigned)((idx >> 9) & 31u);
    unsigned nt   = (unsigned)((idx >> 14) & 127u);
    unsigned l    = (unsigned)(idx >> 21);
    unsigned n    = nt * 16u + (lane & 15u);
    unsigned kl   = ((lane & 16u) ? 8u : 0u) + (q < 8u ? q : (q + 8u));
    unsigned Kg   = kb * 32u + kl;
    float v;
    if (Kg < EMB) v = Wi[(size_t)l * GATES * EMB + (size_t)n * EMB + Kg];
    else          v = Wh[(size_t)l * GATES * HID + (size_t)n * HID + (Kg - EMB)];
    __bf16 hi = (__bf16)v;
    wpH[idx] = hi;
    wpL[idx] = (__bf16)(v - (float)hi);
  }
}

// ---------------------------------------------------------------------------
// Persistent bidirectional scan over all layers.
//   wg: dir = wg&1, bt = (wg>>1)&3 (batch tile), jg = wg>>3 (0..3)
//   wave -> jt = jg*8 + wave (hidden tile 0..31)
// ---------------------------------------------------------------------------
__global__ __launch_bounds__(TPB, 1)
void bilstm_scan_kernel(const float* __restrict__ x,
                        const float* __restrict__ bias,
                        const __bf16* __restrict__ wpH,
                        const __bf16* __restrict__ wpL,
                        __bf16* __restrict__ xinH, __bf16* __restrict__ xinL,
                        float* __restrict__ hsF,  float* __restrict__ hsB,
                        __bf16* __restrict__ hsp,            // [dir][hi/lo][64*512]
                        float* __restrict__ out,
                        unsigned* bar) {
  __shared__ __align__(16) __bf16 As[2][16 * KDIM];   // [hi/lo][row][K] = 64 KB

  const int tid  = threadIdx.x;
  const int wg   = blockIdx.x;
  const int dir  = wg & 1;
  const int id2  = wg >> 1;          // 0..15
  const int bt   = id2 & 3;          // batch tile
  const int jg   = id2 >> 2;         // 0..3
  const int wave = tid >> 5;
  const int lane = tid & 31;
  const int half = lane >> 4;
  const int r    = lane & 15;
  const int jt   = jg * 8 + wave;    // hidden tile 0..31

  __bf16* myHspH = hsp + (size_t)dir * 2 * BATCH * HID;
  __bf16* myHspL = myHspH + (size_t)BATCH * HID;
  float*  myHs   = dir ? hsB : hsF;

  const size_t gstride = (size_t)NWG * TPB;
  const size_t gtid    = (size_t)wg * TPB + tid;

  // ---- split the network input into bf16 hi/lo -----------------------------
  for (size_t i = gtid; i < XIN_ELEMS; i += gstride) {
    float v = x[i];
    __bf16 h = (__bf16)v;
    xinH[i] = h;
    xinL[i] = (__bf16)(v - (float)h);
  }
  gbar(bar);

  for (int l = 0; l < NLAYER; ++l) {
    // h0 = 0 for both directions
    for (size_t i = gtid; i < (size_t)4 * BATCH * HID; i += gstride)
      hsp[i] = (__bf16)0.0f;
    gbar(bar);

    // per-column gate biases (column fixed per lane for the whole layer)
    float bg[4];
#pragma unroll
    for (int g = 0; g < 4; ++g)
      bg[g] = bias[(size_t)l * GATES + (size_t)g * HID + (size_t)jt * 16 + r];

    v8f cacc = {};   // cell-state tile, lives in registers for all 256 steps

    for (int t = 0; t < S_LEN; ++t) {
      const int s = dir ? (S_LEN - 1 - t) : t;

      // ---- stage A = [x_s | h] (hi/lo) into LDS: 4096 uint4 ----------------
      for (int c = tid; c < 4096; c += TPB) {
        int buf = c >> 11;            // 0 = hi, 1 = lo
        int rem = c & 2047;
        int row = rem >> 7;           // 0..15 (128 uint4 per row)
        int col = rem & 127;          // uint4 within row
        int b   = bt * 16 + row;
        uint4 val;
        if (col < 64) {
          const __bf16* src = (buf ? xinL : xinH) + ((size_t)s * BATCH + b) * EMB;
          val = ((const uint4*)src)[col];
        } else {
          const __bf16* src = (buf ? myHspL : myHspH) + (size_t)b * HID;
          val = ((const uint4*)src)[col - 64];
        }
        ((uint4*)&As[buf][0])[(size_t)row * 128 + col] = val;
      }
      __syncthreads();

      // ---- z tiles for the 4 gates, K = 1024, 3-term bf16 split ------------
      v8f acc[4];
#pragma unroll
      for (int g = 0; g < 4; ++g) {
        v8f a;
#pragma unroll
        for (int e = 0; e < 8; ++e) a[e] = bg[g];
        acc[g] = a;
      }

#pragma unroll 2
      for (int kb = 0; kb < KB_CNT; ++kb) {
        // A fragments (ISA 16-bit A layout: lanes 0-15 K {0..7,16..23},
        // lanes 16-31 K {8..15,24..31}, row = lane&15)
        Frag16 aH, aL;
        {
          const __bf16* pH = &As[0][(size_t)r * KDIM + kb * 32 + (half ? 8 : 0)];
          aH.q[0] = *(const uint4*)(pH);
          aH.q[1] = *(const uint4*)(pH + 16);
          const __bf16* pL = &As[1][(size_t)r * KDIM + kb * 32 + (half ? 8 : 0)];
          aL.q[0] = *(const uint4*)(pL);
          aL.q[1] = *(const uint4*)(pL + 16);
        }
#pragma unroll
        for (int g = 0; g < 4; ++g) {
          const int    nt   = g * 32 + jt;
          const size_t boff = ((((size_t)l * NT_CNT + nt) * KB_CNT + kb) * 512) + (size_t)lane * 16;
          Frag16 bH, bL;
          bH.q[0] = ((const uint4*)(wpH + boff))[0];
          bH.q[1] = ((const uint4*)(wpH + boff))[1];
          bL.q[0] = ((const uint4*)(wpL + boff))[0];
          bL.q[1] = ((const uint4*)(wpL + boff))[1];
          acc[g] = __builtin_amdgcn_wmma_f32_16x16x32_bf16(false, aH.v, false, bH.v,
                                                           (short)0, acc[g], false, false);
          acc[g] = __builtin_amdgcn_wmma_f32_16x16x32_bf16(false, aH.v, false, bL.v,
                                                           (short)0, acc[g], false, false);
          acc[g] = __builtin_amdgcn_wmma_f32_16x16x32_bf16(false, aL.v, false, bH.v,
                                                           (short)0, acc[g], false, false);
        }
      }

      // ---- gates + state update (C/D layout: row = e + 8*half, col = r) ----
#pragma unroll
      for (int e = 0; e < 8; ++e) {
        float ig = sigf(acc[0][e]);
        float fg = sigf(acc[1][e]);
        float og = sigf(acc[2][e]);
        float gg = tanhfast(acc[3][e]);
        float c  = fg * cacc[e] + ig * gg;
        cacc[e]  = c;
        float h  = og * tanhfast(c);
        int b = bt * 16 + e + 8 * half;
        int j = jt * 16 + r;
        myHs[((size_t)s * BATCH + b) * HID + j] = h;
        __bf16 hh = (__bf16)h;
        myHspH[(size_t)b * HID + j] = hh;
        myHspL[(size_t)b * HID + j] = (__bf16)(h - (float)hh);
      }
      __threadfence();
      gbar(bar);
    } // t

    // ---- combine directions: next-layer input split, or final outputs ------
    if (l == NLAYER - 1) {
      for (size_t i = gtid; i < HS_ELEMS; i += gstride)
        out[i] = hsF[i] + hsB[i];
      for (size_t i = gtid; i < (size_t)BATCH * HID; i += gstride)
        out[HS_ELEMS + i] = hsF[(size_t)(S_LEN - 1) * BATCH * HID + i] + hsB[i];
    } else {
      for (size_t i = gtid; i < HS_ELEMS; i += gstride) {
        float v = hsF[i] + hsB[i];
        __bf16 h = (__bf16)v;
        xinH[i] = h;
        xinL[i] = (__bf16)(v - (float)h);
      }
      gbar(bar);
    }
  } // layer
}

// ---------------------------------------------------------------------------
extern "C" void kernel_launch(void* const* d_in, const int* in_sizes, int n_in,
                              void* d_out, int out_size, void* d_ws, size_t ws_size,
                              hipStream_t stream) {
  const float* x  = (const float*)d_in[0];   // [S,B,E]
  const float* Wi = (const float*)d_in[1];   // [L,4H,E]
  const float* Wh = (const float*)d_in[2];   // [L,4H,H]
  const float* b  = (const float*)d_in[3];   // [L,4H]

  char* ws = (char*)d_ws;
  __bf16*   wpH  = (__bf16*)(ws + OFF_WP_HI);
  __bf16*   wpL  = (__bf16*)(ws + OFF_WP_LO);
  __bf16*   xinH = (__bf16*)(ws + OFF_XIN_HI);
  __bf16*   xinL = (__bf16*)(ws + OFF_XIN_LO);
  float*    hsF  = (float*)(ws + OFF_HS_F);
  float*    hsB  = (float*)(ws + OFF_HS_B);
  __bf16*   hsp  = (__bf16*)(ws + OFF_HSP);
  unsigned* bar  = (unsigned*)(ws + OFF_BAR);

  hipLaunchKernelGGL(init_bar_kernel, dim3(1), dim3(1), 0, stream, bar);
  hipLaunchKernelGGL(prep_weights_kernel, dim3(8192), dim3(TPB), 0, stream, Wi, Wh, wpH, wpL);
  hipLaunchKernelGGL(bilstm_scan_kernel, dim3(NWG), dim3(TPB), 0, stream,
                     x, b, wpH, wpL, xinH, xinL, hsF, hsB, hsp, (float*)d_out, bar);
}
